// SNS_Control_47390669144758
// MI455X (gfx1250) — matrix operand order repaired and checked by reading, not compiled, not run on hardware
//
#include <hip/hip_runtime.h>

// MI455X / gfx1250, wave32. One wave = one 16-row batch tile of the LTC scan.
// Per step & cell: A = [g(sensory inputs) | g(recurrent state)] (16x32 f16, K padded),
// two v_wmma_f32_16x16x32_f16 (num = w*erev weights, den = w weights), then the
// elementwise semi-implicit Euler update in the WMMA D layout.
// Single-wave blocks: intra-wave LDS ordering is guaranteed in-order by HW, so we
// use a compiler-only wave barrier (no s_barrier, no loadcnt stall) so the t+1
// global-input prefetch overlaps the whole WMMA/update section.

#define DT_INV 240.0f

typedef _Float16 v16h __attribute__((ext_vector_type(16)));
typedef _Float16 v8h  __attribute__((ext_vector_type(8)));
typedef float    v8f  __attribute__((ext_vector_type(8)));

struct CellP {
  const float *tau, *b, *in_w, *in_b, *out_w, *out_b;
  const float *smu, *ssig, *serev, *sw;
  const float *mu, *sig, *erev, *w;
};

__device__ __forceinline__ void wavebar(){
  // Compiler ordering fence only. Block == one wave32, and same-wave LDS ops
  // are architecturally in-order (DScnt, "LDS: in order"), so no s_barrier and
  // no global-load wait is required between the LDS scatter and the A-gather.
  __builtin_amdgcn_wave_barrier();
}

__device__ __forceinline__ float fast_rcp(float d){
  float r = __builtin_amdgcn_rcpf(d);
  return r * (2.0f - d * r);          // one Newton iteration -> ~full f32 precision
}
__device__ __forceinline__ _Float16 gact(float u, float sig, float mu){
  float z = sig * (u - mu);
  float e = __expf(-z);
  return (_Float16)__builtin_amdgcn_rcpf(1.0f + e);
}

// Build this lane's slice of the 32x16 f16 B operand.
// Rows 0..I-1 = sensory weights, rows I..I+N-1 = recurrent weights, rest 0.
// Lane layout: lanes 0-15 hold K=0..7,16..23 (col n=lane), lanes 16-31 hold K=8..15,24..31.
__device__ __forceinline__ v16h buildB(const CellP& p, int I, int N, bool num,
                                       int n, int sel){
  v16h bb;
  #pragma unroll
  for (int h = 0; h < 16; ++h){
    int k = (h < 8) ? (8 * sel + h) : (16 + 8 * sel + (h - 8));
    float val = 0.0f;
    if (n < N){
      if (k < I){
        float wv = p.sw[k * N + n];
        val = num ? wv * p.serev[k * N + n] : wv;
      } else if (k < I + N){
        int i = k - I;
        float wv = p.w[i * N + n];
        val = num ? wv * p.erev[i * N + n] : wv;
      }
    }
    bb[h] = (_Float16)val;
  }
  return bb;
}

__global__ __launch_bounds__(32) void sns_scan_kernel(
    const float* __restrict__ inp, const float* __restrict__ obj,
    const float* __restrict__ tgt, float* __restrict__ out,
    CellP p1, CellP p2, CellP p3, int T)
{
  __shared__ __align__(16) _Float16 sA1[16 * 32];
  __shared__ __align__(16) _Float16 sA2[16 * 32];
  __shared__ __align__(16) _Float16 sA3[16 * 32];

  const int lane = threadIdx.x;      // 0..31
  const int j    = lane & 15;        // D-layout column / B-layout column
  const int sel  = lane >> 4;        // lane-group
  const int b0   = blockIdx.x * 16;  // batch tile origin

  // Zero A tiles once: padded K slots must stay zero forever.
  for (int idx = lane; idx < 512; idx += 32){
    sA1[idx] = (_Float16)0.0f; sA2[idx] = (_Float16)0.0f; sA3[idx] = (_Float16)0.0f;
  }

  // Constant B operands (built once, live in VGPRs for the whole scan).
  const v16h B1n = buildB(p1,  8,  4, true,  j, sel);
  const v16h B1d = buildB(p1,  8,  4, false, j, sel);
  const v16h B2n = buildB(p2, 12, 10, true,  j, sel);
  const v16h B2d = buildB(p2, 12, 10, false, j, sel);
  const v16h B3n = buildB(p3, 12,  5, true,  j, sel);
  const v16h B3d = buildB(p3, 12,  5, false, j, sel);

  // ---- per-lane constants --------------------------------------------------
  // Cell-1 sensory loader: lane handles row xrow, input cols c0..c0+3.
  const int xrow = lane >> 1;
  const int c0   = (lane & 1) * 4;
  float inw1[4], inb1[4], ss1[4], sm1[4];
  #pragma unroll
  for (int c = 0; c < 4; ++c){
    inw1[c] = p1.in_w[c0 + c]; inb1[c] = p1.in_b[c0 + c];
    ss1[c]  = p1.ssig[(c0 + c) * 4]; sm1[c] = p1.smu[(c0 + c) * 4];
  }
  // obj/tgt feeders for cell-2: lanes<16 -> o_r (k=0..4), lanes>=16 -> t_r (k=5..9).
  const int    otrow = j;
  const int    kbase = sel ? 5 : 0;
  const float* otsrc = sel ? tgt : obj;
  float inw2o[5], inb2o[5], ss2o[5], sm2o[5];
  #pragma unroll
  for (int c = 0; c < 5; ++c){
    int k = kbase + c;
    inw2o[c] = p2.in_w[k]; inb2o[c] = p2.in_b[k];
    ss2o[c]  = p2.ssig[k * 10]; sm2o[c] = p2.smu[k * 10];
  }
  // State-update / recurrent-g constants per output column j.
  // For invalid columns keep zeros: with zeroed B columns the unconditional
  // update yields (0+0+0)*rcp(1) = 0 (no NaN), so updates run on all lanes.
  float cmt1 = 0, bb1 = 0, rs1 = 0, rm1 = 0;
  if (j < 4){ cmt1 = p1.tau[j] * DT_INV; bb1 = p1.b[j];
              rs1 = p1.sig[j * 4]; rm1 = p1.mu[j * 4]; }
  float cmt2 = 0, bb2 = 0, rs2 = 0, rm2 = 0, inw3j = 0, inb3j = 0, ss3j = 0, sm3j = 0;
  if (j < 10){ cmt2 = p2.tau[j] * DT_INV; bb2 = p2.b[j];
               rs2 = p2.sig[j * 10]; rm2 = p2.mu[j * 10];
               inw3j = p3.in_w[j]; inb3j = p3.in_b[j];
               ss3j = p3.ssig[j * 5]; sm3j = p3.smu[j * 5]; }
  float cmt3 = 0, bb3 = 0, rs3 = 0, rm3 = 0, ow3 = 0, ob3 = 0;
  if (j < 5){ cmt3 = p3.tau[j] * DT_INV; bb3 = p3.b[j];
              rs3 = p3.sig[j * 5]; rm3 = p3.mu[j * 5];
              ow3 = p3.out_w[j]; ob3 = p3.out_b[j]; }
  // v1 -> cell2 (k=10,11) and v1 -> cell3 (k=10,11) feeder constants.
  float inw2v = 0, inb2v = 0, ss2v = 0, sm2v = 0;
  if (j < 2){ inw2v = p2.in_w[10 + j]; inb2v = p2.in_b[10 + j];
              ss2v = p2.ssig[(10 + j) * 10]; sm2v = p2.smu[(10 + j) * 10]; }
  float inw3v = 0, inb3v = 0, ss3v = 0, sm3v = 0;
  if (j >= 2 && j < 4){
    int k = 10 + (j - 2);
    inw3v = p3.in_w[k]; inb3v = p3.in_b[k];
    ss3v = p3.ssig[k * 5]; sm3v = p3.smu[k * 5];
  }

  // ---- carried state in WMMA D layout (vgpr r <-> batch row, lane <-> col j)
  float v1[8], v2[8], v3[8];
  const float v3init = (j == 2) ? 0.0f : 10.0f;   // [R/2, R/2, 0, R/2, R/2]
  #pragma unroll
  for (int r = 0; r < 8; ++r){ v1[r] = 0.0f; v2[r] = 0.0f; v3[r] = v3init; }

  // Preload t=0 global inputs.
  float4 xc = *(const float4*)(inp + ((size_t)b0 + xrow) * 8 + c0);
  const float* otp0 = otsrc + ((size_t)b0 + otrow) * 3;
  float oc0 = otp0[0], oc1 = otp0[1], oc2 = otp0[2];

  const v8f zz = {};

  for (int t = 0; t < T; ++t){
    // -- recurrent A slots from carried state (next K block of each tile)
    if (j < 4){
      #pragma unroll
      for (int r = 0; r < 8; ++r)
        sA1[(r + 8 * sel) * 32 + 8 + j] = gact(v1[r], rs1, rm1);
    }
    if (j < 10){
      #pragma unroll
      for (int r = 0; r < 8; ++r)
        sA2[(r + 8 * sel) * 32 + 12 + j] = gact(v2[r], rs2, rm2);
    }
    if (j < 5){
      #pragma unroll
      for (int r = 0; r < 8; ++r)
        sA3[(r + 8 * sel) * 32 + 12 + j] = gact(v3[r], rs3, rm3);
    }
    // -- cell-1 sensory slots from x_t
    {
      float xv[4] = {xc.x, xc.y, xc.z, xc.w};
      #pragma unroll
      for (int c = 0; c < 4; ++c){
        float u = inw1[c] * xv[c] + inb1[c];
        sA1[xrow * 32 + c0 + c] = gact(u, ss1[c], sm1[c]);
      }
    }
    // -- cell-2 sensory slots from o_r / t_r (repeat(·,2)[:, :-1] pattern)
    {
      float ov[3] = {oc0, oc1, oc2};
      const int src[5] = {0, 0, 1, 1, 2};
      #pragma unroll
      for (int c = 0; c < 5; ++c){
        float u = inw2o[c] * ov[src[c]] + inb2o[c];
        sA2[otrow * 32 + kbase + c] = gact(u, ss2o[c], sm2o[c]);
      }
    }
    // -- prefetch next step's inputs; with wave-only barriers below these
    //    loads stay in flight across the entire WMMA/update section.
    if (t + 1 < T){
      xc = *(const float4*)(inp + ((size_t)(t + 1) * 2048 + b0 + xrow) * 8 + c0);
      const float* op = otsrc + ((size_t)(t + 1) * 2048 + b0 + otrow) * 3;
      oc0 = op[0]; oc1 = op[1]; oc2 = op[2];
    }
    wavebar();

    union AV { v16h v; v8h h[2]; } a;
    // ================= cell 1 =================
    a.h[0] = *reinterpret_cast<const v8h*>(&sA1[j * 32 + sel * 8]);
    a.h[1] = *reinterpret_cast<const v8h*>(&sA1[j * 32 + 16 + sel * 8]);
    v8f n1 = __builtin_amdgcn_wmma_f32_16x16x32_f16(false, a.v, false, B1n, (short)0, zz, false, false);
    v8f d1 = __builtin_amdgcn_wmma_f32_16x16x32_f16(false, a.v, false, B1d, (short)0, zz, false, false);
    #pragma unroll
    for (int r = 0; r < 8; ++r)
      v1[r] = (cmt1 * v1[r] + bb1 + n1[r]) * fast_rcp(cmt1 + 1.0f + d1[r]);
    // feed v1_new into cell-2 (cols 0,1) and cell-3 (cols 2,3) sensory slots
    if (j < 2){
      #pragma unroll
      for (int r = 0; r < 8; ++r)
        sA2[(r + 8 * sel) * 32 + 10 + j] = gact(inw2v * v1[r] + inb2v, ss2v, sm2v);
    } else if (j < 4){
      #pragma unroll
      for (int r = 0; r < 8; ++r)
        sA3[(r + 8 * sel) * 32 + 10 + (j - 2)] = gact(inw3v * v1[r] + inb3v, ss3v, sm3v);
    }
    wavebar();

    // ================= cell 2 =================
    a.h[0] = *reinterpret_cast<const v8h*>(&sA2[j * 32 + sel * 8]);
    a.h[1] = *reinterpret_cast<const v8h*>(&sA2[j * 32 + 16 + sel * 8]);
    v8f n2 = __builtin_amdgcn_wmma_f32_16x16x32_f16(false, a.v, false, B2n, (short)0, zz, false, false);
    v8f d2 = __builtin_amdgcn_wmma_f32_16x16x32_f16(false, a.v, false, B2d, (short)0, zz, false, false);
    #pragma unroll
    for (int r = 0; r < 8; ++r)
      v2[r] = (cmt2 * v2[r] + bb2 + n2[r]) * fast_rcp(cmt2 + 1.0f + d2[r]);
    if (j < 10){
      #pragma unroll
      for (int r = 0; r < 8; ++r)
        sA3[(r + 8 * sel) * 32 + j] = gact(inw3j * v2[r] + inb3j, ss3j, sm3j);
    }
    wavebar();

    // ================= cell 3 =================
    a.h[0] = *reinterpret_cast<const v8h*>(&sA3[j * 32 + sel * 8]);
    a.h[1] = *reinterpret_cast<const v8h*>(&sA3[j * 32 + 16 + sel * 8]);
    v8f n3 = __builtin_amdgcn_wmma_f32_16x16x32_f16(false, a.v, false, B3n, (short)0, zz, false, false);
    v8f d3 = __builtin_amdgcn_wmma_f32_16x16x32_f16(false, a.v, false, B3d, (short)0, zz, false, false);
    #pragma unroll
    for (int r = 0; r < 8; ++r)
      v3[r] = (cmt3 * v3[r] + bb3 + n3[r]) * fast_rcp(cmt3 + 1.0f + d3[r]);
    if (j < 5){
      #pragma unroll
      for (int r = 0; r < 8; ++r)
        out[((size_t)t * 2048 + b0 + r + 8 * sel) * 5 + j] = v3[r] * ow3 + ob3;
    }
    wavebar();
  }
}

extern "C" void kernel_launch(void* const* d_in, const int* in_sizes, int n_in,
                              void* d_out, int out_size, void* d_ws, size_t ws_size,
                              hipStream_t stream)
{
  const float* inp = (const float*)d_in[0];
  const float* obj = (const float*)d_in[1];
  const float* tgt = (const float*)d_in[2];

  auto getP = [&](int base) -> CellP {
    CellP p;
    p.tau   = (const float*)d_in[base + 0];
    p.b     = (const float*)d_in[base + 1];
    p.in_w  = (const float*)d_in[base + 2];
    p.in_b  = (const float*)d_in[base + 3];
    p.out_w = (const float*)d_in[base + 4];
    p.out_b = (const float*)d_in[base + 5];
    p.smu   = (const float*)d_in[base + 6];
    p.ssig  = (const float*)d_in[base + 7];
    p.serev = (const float*)d_in[base + 8];
    p.sw    = (const float*)d_in[base + 9];
    p.mu    = (const float*)d_in[base + 10];
    p.sig   = (const float*)d_in[base + 11];
    p.erev  = (const float*)d_in[base + 12];
    p.w     = (const float*)d_in[base + 13];
    return p;
  };
  CellP p1 = getP(3), p2 = getP(17), p3 = getP(31);

  const int T = in_sizes[0] / (2048 * 8);   // (T, 2048, 8) f32
  dim3 grid(2048 / 16), block(32);
  hipLaunchKernelGGL(sns_scan_kernel, grid, block, 0, stream,
                     inp, obj, tgt, (float*)d_out, p1, p2, p3, T);
}
